// ElmoLstmUni_42133629173986
// MI455X (gfx1250) — matrix operand order repaired and checked by reading, not compiled, not run on hardware
//
#include <hip/hip_runtime.h>
#include <cstdint>
#include <cstddef>

// ---------------------------------------------------------------------------
// ElmoLstm (2-layer LSTM with projection, ELMo residual) for MI455X (gfx1250)
//
// bf16 WMMA 16x16x32 with fp32 accumulate; weights pre-converted to bf16 so
// both layers (~75MB) stay resident in the 192MB L2. The sequence is a serial
// chain of 2048 steps, so kernels are shaped for minimum critical path:
// wide grids, few WMMAs per wave, K-split + LDS reduction in the projection,
// async global->LDS DMA for activation staging, WGP-scope weight prefetch.
// ---------------------------------------------------------------------------

typedef __bf16 bf16_t;
typedef __attribute__((ext_vector_type(16))) __bf16 v16bf;
typedef __attribute__((ext_vector_type(8)))  float  v8f;

union Frag16 { v16bf v; uint4 q[2]; };   // 16 bf16 = 32B = two 16B chunks

constexpr int kB = 32;      // batch
constexpr int kT = 1024;    // time
constexpr int kH = 512;     // hidden / projection
constexpr int kC = 4096;    // cell
constexpr int kG = 4 * kC;  // 16384 gate rows
constexpr int kK = 2 * kH;  // 1024 = concat [x | h] K-dim
constexpr int kL = 2;       // layers
constexpr float kClip = 3.0f;

__device__ __forceinline__ v8f wmma_bf16(v16bf a, v16bf b, v8f c) {
  return __builtin_amdgcn_wmma_f32_16x16x32_bf16(
      false, a, false, b, (short)0, c, false, false);
}

__device__ __forceinline__ float sigmoidf_fast(float x) {
  return 1.0f / (1.0f + __expf(-x));
}

// WGP-scope prefetch (scope 0 pulls into all cache levels; the builtin only
// reaches SE/SYS scope which stop at GL2 -- useless for L2-resident weights).
__device__ __forceinline__ void prefetch_wgp(const void* p) {
  asm volatile("global_prefetch_b8 %0, off" :: "v"(p));
}

// Async DMA: 16B per lane from global into LDS (per-lane LDS scatter address),
// tracked with ASYNCcnt.
__device__ __forceinline__ void async_g2lds_b128(uint32_t lds_byte_off,
                                                 const void* gsrc) {
  asm volatile("global_load_async_to_lds_b128 %0, %1, off"
               :: "v"(lds_byte_off), "v"(gsrc) : "memory");
}
__device__ __forceinline__ void wait_async0() {
  asm volatile("s_wait_asynccnt 0x0" ::: "memory");
}

// ---------------------------------------------------------------------------
// Kernel A: gates GEMM + LSTM cell. Grid: kC/32 = 128 WGs, 256 threads.
// WG owns 32 cell indices -> 128 gate columns (i/f/g/o for those cells).
// Wave w: gate = w>>1, N-tile = w&1 -> 16 columns, 2 M-tiles, K=1024.
// Critical path: 64 WMMAs/wave.
// ---------------------------------------------------------------------------
__global__ __launch_bounds__(256, 1) void k_gates_cell(
    const bf16_t* __restrict__ seq,    // [B,T,H] bf16 layer input
    const bf16_t* __restrict__ h16,    // [B,H]   bf16 h_{t-1}
    const bf16_t* __restrict__ wcat,   // [kG,kK] bf16 [Wi|Ws]
    const float*  __restrict__ bias,   // [kG]
    const int*    __restrict__ len,    // [B]
    float*        __restrict__ c,      // [B,kC]  cell state (in/out)
    bf16_t*       __restrict__ a,      // [B,kC]  o*tanh(c) out (bf16)
    int t) {
  __shared__ char smem[65536];         // single static alloc => LDS offset 0
  bf16_t* xh   = reinterpret_cast<bf16_t*>(smem);  // [32][1024] bf16, swizzled
  float*  gbuf = reinterpret_cast<float*>(smem);   // reused: [4][32][32] f32

  const int tid = threadIdx.x;

  // Stage A = [x_t | h_prev] into LDS via async DMA (no VGPR round trip).
  // XOR-swizzle 16B chunks by (m&15) to spread the 2KB row stride across banks.
  for (int i = 0; i < 16; ++i) {
    int cidx = tid + i * 256;          // 4096 chunks of 16B
    int m    = cidx >> 7;
    int ch   = cidx & 127;
    int off  = ch << 3;
    const void* src;
    if (off < kH)
      src = (const void*)(seq + ((size_t)m * kT + t) * kH + off);
    else
      src = (const void*)(h16 + (size_t)m * kH + (off - kH));
    int chs = ch ^ (m & 15);
    uint32_t loff = (uint32_t)((m * kK + (chs << 3)) * sizeof(bf16_t));
    async_g2lds_b128(loff, src);
  }
  wait_async0();
  __syncthreads();

  const int lane  = tid & 31;
  const int w     = tid >> 5;
  const int gate  = w >> 1;                 // 0..3 : i,f,g,o
  const int half  = w & 1;                  // N-tile within the 32-cell chunk
  const int j0    = blockIdx.x * 32;        // cell-index chunk base
  const int nrow0 = gate * kC + j0 + half * 16;
  const int ln    = lane & 15;
  const int hi    = lane >> 4;

  const v8f vzero = {0.f,0.f,0.f,0.f,0.f,0.f,0.f,0.f};
  v8f acc0 = vzero, acc1 = vzero;

  for (int k0 = 0; k0 < kK; k0 += 32) {
    // A fragments (16-bit A layout: lanes 0-15 K{0-7,16-23}, 16-31 K{8-15,24-31})
    Frag16 A0, A1, Bf;
    int c0 = (k0 >> 3) + hi;
    int m0 = ln, m1 = 16 + ln;
    A0.q[0] = *reinterpret_cast<const uint4*>(xh + m0 * kK + (((c0    ) ^ (m0 & 15)) << 3));
    A0.q[1] = *reinterpret_cast<const uint4*>(xh + m0 * kK + (((c0 + 2) ^ (m0 & 15)) << 3));
    A1.q[0] = *reinterpret_cast<const uint4*>(xh + m1 * kK + (((c0    ) ^ (m1 & 15)) << 3));
    A1.q[1] = *reinterpret_cast<const uint4*>(xh + m1 * kK + (((c0 + 2) ^ (m1 & 15)) << 3));
    // B fragment: lane ln = column, K contiguous in the [N,K] weight row.
    const bf16_t* brow = wcat + (size_t)(nrow0 + ln) * kK + k0 + hi * 8;
    Bf.q[0] = *reinterpret_cast<const uint4*>(brow);
    Bf.q[1] = *reinterpret_cast<const uint4*>(brow + 16);
    prefetch_wgp(brow + 32);               // next K step into WGP cache
    acc0 = wmma_bf16(A0.v, Bf.v, acc0);
    acc1 = wmma_bf16(A1.v, Bf.v, acc1);
  }

  __syncthreads();   // xh dead; reuse LDS as the gate exchange buffer

  // C/D layout: element r of v8f -> M = r + (hi?8:0), N = ln
#pragma unroll
  for (int mt = 0; mt < 2; ++mt) {
    v8f accv = mt ? acc1 : acc0;
#pragma unroll
    for (int r = 0; r < 8; ++r) {
      int mrow = mt * 16 + r + hi * 8;
      int col  = half * 16 + ln;
      gbuf[(gate * 32 + mrow) * 32 + col] = accv[r];
    }
  }
  __syncthreads();

  // Elementwise LSTM cell for this WG's 32x32 cells (4 per thread).
#pragma unroll
  for (int i = 0; i < 4; ++i) {
    int e   = tid + i * 256;   // 0..1023
    int m   = e >> 5;
    int col = e & 31;
    int j   = j0 + col;
    float ig = gbuf[(0 * 32 + m) * 32 + col] + bias[0 * kC + j];
    float fg = gbuf[(1 * 32 + m) * 32 + col] + bias[1 * kC + j];
    float mg = gbuf[(2 * 32 + m) * 32 + col] + bias[2 * kC + j];
    float og = gbuf[(3 * 32 + m) * 32 + col] + bias[3 * kC + j];
    float iv = sigmoidf_fast(ig);
    float fv = sigmoidf_fast(fg);
    float gv = tanhf(mg);
    float ov = sigmoidf_fast(og);
    size_t idx = (size_t)m * kC + j;
    float cp = c[idx];
    float cn = fminf(fmaxf(iv * gv + fv * cp, -kClip), kClip);
    bool valid = (t < len[m]);
    c[idx] = valid ? cn : cp;
    a[idx] = (bf16_t)(ov * tanhf(cn));   // masked at the h stage
  }
}

// ---------------------------------------------------------------------------
// Kernel B: projection h = clip((o*tanh(c)) @ Wp^T), K-split 4 ways.
// Grid: kH/32 = 16 WGs, 256 threads. Wave w: N-tile = w>>2, K-split = w&3
// (each wave covers K range of 1024 -> 64 WMMAs). Partials reduced in LDS.
// ---------------------------------------------------------------------------
__global__ __launch_bounds__(256, 1) void k_proj(
    const bf16_t* __restrict__ a,      // [B,kC] bf16
    const bf16_t* __restrict__ wp,     // [kH,kC] bf16 ([N,K])
    const int*    __restrict__ len,
    float*        __restrict__ h32,    // [B,kH] fp32 state (in/out)
    bf16_t*       __restrict__ h16,    // [B,kH] bf16 copy for next step A
    float*        __restrict__ out0,   // [B,T,kH] layer-0 masked output
    bf16_t*       __restrict__ seq1,   // [B,T,kH] bf16 layer-1 input
    float*        __restrict__ outret, // [B,T,kH] final returned output
    int t, int layer) {
  __shared__ float psum[2 * 4 * 32 * 16];   // [nt][ks][M=32][N=16], 16KB

  const int tid = threadIdx.x, lane = tid & 31, w = tid >> 5;
  const int nt = w >> 2;                    // 0/1 N-tile in this WG
  const int ks = w & 3;                     // K split 0..3
  const int n0 = blockIdx.x * 32 + nt * 16;
  const int kbase = ks * (kC / 4);          // 1024-wide K slice
  const int ln = lane & 15, hi = lane >> 4;

  const v8f vzero = {0.f,0.f,0.f,0.f,0.f,0.f,0.f,0.f};
  v8f acc0 = vzero, acc1 = vzero;
  const bf16_t* ar0 = a + (size_t)(ln)      * kC + kbase;
  const bf16_t* ar1 = a + (size_t)(16 + ln) * kC + kbase;
  const bf16_t* br  = wp + (size_t)(n0 + ln) * kC + kbase + hi * 8;

  for (int k0 = 0; k0 < kC / 4; k0 += 32) {
    Frag16 A0, A1, Bf;
    A0.q[0] = *reinterpret_cast<const uint4*>(ar0 + k0 + hi * 8);
    A0.q[1] = *reinterpret_cast<const uint4*>(ar0 + k0 + 16 + hi * 8);
    A1.q[0] = *reinterpret_cast<const uint4*>(ar1 + k0 + hi * 8);
    A1.q[1] = *reinterpret_cast<const uint4*>(ar1 + k0 + 16 + hi * 8);
    Bf.q[0] = *reinterpret_cast<const uint4*>(br + k0);
    Bf.q[1] = *reinterpret_cast<const uint4*>(br + k0 + 16);
    prefetch_wgp(br + k0 + 32);
    acc0 = wmma_bf16(A0.v, Bf.v, acc0);
    acc1 = wmma_bf16(A1.v, Bf.v, acc1);
  }

  // Write partials: element r -> M = mt*16 + r + hi*8, N = ln
#pragma unroll
  for (int mt = 0; mt < 2; ++mt) {
    v8f accv = mt ? acc1 : acc0;
#pragma unroll
    for (int r = 0; r < 8; ++r) {
      int m = mt * 16 + r + hi * 8;
      psum[((nt * 4 + ks) * 32 + m) * 16 + ln] = accv[r];
    }
  }
  __syncthreads();

  // Reduce 4 K-partials + epilogue: 2*32*16 = 1024 outputs, 4 per thread.
#pragma unroll
  for (int i = 0; i < 4; ++i) {
    int e  = tid + i * 256;       // 0..1023
    int et = e >> 9;              // nt
    int rm = e & 511;
    int m  = rm >> 4;
    int nc = rm & 15;
    int n  = blockIdx.x * 32 + et * 16 + nc;
    float s = psum[((et * 4 + 0) * 32 + m) * 16 + nc]
            + psum[((et * 4 + 1) * 32 + m) * 16 + nc]
            + psum[((et * 4 + 2) * 32 + m) * 16 + nc]
            + psum[((et * 4 + 3) * 32 + m) * 16 + nc];
    float h = fminf(fmaxf(s, -kClip), kClip);
    bool valid = (t < len[m]);
    size_t sidx = (size_t)m * kH + n;
    float hp = h32[sidx];
    float hn = valid ? h : hp;
    h32[sidx] = hn;
    h16[sidx] = (bf16_t)hn;
    float ov = valid ? h : 0.f;
    size_t oidx = ((size_t)m * kT + t) * kH + n;
    if (layer == 0) {
      out0[oidx] = ov;                 // outs[0] (also residual source)
      seq1[oidx] = (bf16_t)ov;         // layer-1 input
    } else {
      outret[oidx] = ov + out0[oidx];  // ELMo residual: outs[1] = h + outs[0]
    }
  }
}

// ---------------------------------------------------------------------------
// Prep / utility kernels
// ---------------------------------------------------------------------------
__global__ void k_lengths(const int* __restrict__ mask, int* __restrict__ len) {
  int b = threadIdx.x;
  if (b < kB) {
    int s = 0;
    for (int t = 0; t < kT; ++t) s += mask[b * kT + t];
    len[b] = s;
  }
}

__global__ void k_cvt_bf16(const float* __restrict__ src, bf16_t* __restrict__ dst,
                           size_t n) {
  size_t stride = (size_t)gridDim.x * blockDim.x;
  for (size_t i = blockIdx.x * (size_t)blockDim.x + threadIdx.x; i < n; i += stride)
    dst[i] = (bf16_t)src[i];
}

// Wcat[l][n][0:512] = Wi[l][n][:], Wcat[l][n][512:1024] = Ws[l][n][:]
__global__ void k_cvt_wcat(const float* __restrict__ wi, const float* __restrict__ ws,
                           bf16_t* __restrict__ wcat) {
  const size_t total = (size_t)kL * kG * kK;
  size_t stride = (size_t)gridDim.x * blockDim.x;
  for (size_t idx = blockIdx.x * (size_t)blockDim.x + threadIdx.x; idx < total;
       idx += stride) {
    size_t l   = idx / ((size_t)kG * kK);
    size_t rem = idx % ((size_t)kG * kK);
    size_t n = rem / kK, k = rem % kK;
    float v = (k < kH) ? wi[(l * kG + n) * kH + k]
                       : ws[(l * kG + n) * kH + (k - kH)];
    wcat[idx] = (bf16_t)v;
  }
}

__global__ void k_init_state(float* __restrict__ h32, bf16_t* __restrict__ h16,
                             float* __restrict__ c) {
  int i = blockIdx.x * blockDim.x + threadIdx.x;
  if (i < kB * kH) { h32[i] = 0.f; h16[i] = (bf16_t)0.f; }
  if (i < kB * kC) c[i] = 0.f;
}

// ---------------------------------------------------------------------------
// Host launcher
// ---------------------------------------------------------------------------
extern "C" void kernel_launch(void* const* d_in, const int* in_sizes, int n_in,
                              void* d_out, int out_size, void* d_ws, size_t ws_size,
                              hipStream_t stream) {
  (void)in_sizes; (void)n_in; (void)out_size; (void)ws_size;
  const float* inputs = (const float*)d_in[0];
  const int*   mask   = (const int*)d_in[1];
  const float* W_i    = (const float*)d_in[2];
  const float* W_s    = (const float*)d_in[3];
  const float* b_s    = (const float*)d_in[4];
  const float* W_p    = (const float*)d_in[5];
  float* out = (float*)d_out;

  // Workspace carve-out (~211 MB), 256B-aligned regions.
  char* ws = (char*)d_ws;
  size_t cur = 0;
  auto alloc = [&](size_t bytes) -> char* {
    cur = (cur + 255) & ~(size_t)255;
    char* p = ws + cur;
    cur += bytes;
    return p;
  };
  bf16_t* wcat = (bf16_t*)alloc((size_t)kL * kG * kK * sizeof(bf16_t)); // 64 MB
  bf16_t* wp   = (bf16_t*)alloc((size_t)kL * kH * kC * sizeof(bf16_t)); //  8 MB
  bf16_t* seq0 = (bf16_t*)alloc((size_t)kB * kT * kH * sizeof(bf16_t)); // 32 MB
  bf16_t* seq1 = (bf16_t*)alloc((size_t)kB * kT * kH * sizeof(bf16_t)); // 32 MB
  bf16_t* abuf = (bf16_t*)alloc((size_t)kB * kC * sizeof(bf16_t));
  bf16_t* h16  = (bf16_t*)alloc((size_t)kB * kH * sizeof(bf16_t));
  float*  out0 = (float*) alloc((size_t)kB * kT * kH * sizeof(float));  // 64 MB
  float*  cbuf = (float*) alloc((size_t)kB * kC * sizeof(float));
  float*  h32  = (float*) alloc((size_t)kB * kH * sizeof(float));
  int*    len  = (int*)   alloc(kB * sizeof(int));

  // One-time prep (pure functions of inputs; deterministic).
  k_lengths<<<1, 32, 0, stream>>>(mask, len);
  k_cvt_wcat<<<2048, 256, 0, stream>>>(W_i, W_s, wcat);
  k_cvt_bf16<<<1024, 256, 0, stream>>>(W_p, wp, (size_t)kL * kH * kC);
  k_cvt_bf16<<<1024, 256, 0, stream>>>(inputs, seq0, (size_t)kB * kT * kH);

  const size_t retN = (size_t)kB * kT * kH;

  for (int l = 0; l < kL; ++l) {
    const bf16_t* seq_in = (l == 0) ? seq0 : seq1;
    const bf16_t* wcat_l = wcat + (size_t)l * kG * kK;
    const bf16_t* wp_l   = wp   + (size_t)l * kH * kC;
    const float*  bias_l = b_s  + (size_t)l * kG;

    k_init_state<<<(kB * kC + 255) / 256, 256, 0, stream>>>(h32, h16, cbuf);

    for (int t = 0; t < kT; ++t) {
      k_gates_cell<<<kC / 32, 256, 0, stream>>>(seq_in, h16, wcat_l, bias_l,
                                                len, cbuf, abuf, t);
      k_proj<<<kH / 32, 256, 0, stream>>>(abuf, wp_l, len, h32, h16,
                                          out0, seq1, out, t, l);
    }

    // d_out layout: returned [B,T,H] | final_h [L,B,H] | final_c [L,B,C]
    hipMemcpyAsync(out + retN + (size_t)l * kB * kH, h32,
                   (size_t)kB * kH * sizeof(float), hipMemcpyDeviceToDevice, stream);
    hipMemcpyAsync(out + retN + (size_t)kL * kB * kH + (size_t)l * kB * kC, cbuf,
                   (size_t)kB * kC * sizeof(float), hipMemcpyDeviceToDevice, stream);
  }
}
